// SyncBidirMultiheadAttentionMechanism_72430328480556
// MI455X (gfx1250) — matrix operand order repaired and checked by reading, not compiled, not run on hardware
//
#include <hip/hip_runtime.h>
#include <hip/hip_bf16.h>
#include <math.h>

// ---------------------------------------------------------------------------
// CDNA5 (gfx1250) bf16 WMMA types / helpers
// ---------------------------------------------------------------------------
typedef __bf16 bf16;
typedef __attribute__((ext_vector_type(16))) __bf16 v16bf;
typedef __attribute__((ext_vector_type(8)))  __bf16 v8bf;
typedef __attribute__((ext_vector_type(8)))  float  v8f;

#define NEG_BIG (-3.0e38f)

__device__ __forceinline__ v8f wmma_bf16(v16bf a, v16bf b, v8f c) {
    // D = A(16x32 bf16) * B(32x16 bf16) + C(16x16 f32)
    return __builtin_amdgcn_wmma_f32_16x16x32_bf16(
        /*neg_a=*/false, a, /*neg_b=*/false, b,
        /*c_mod=*/(short)0, c, /*reuse_a=*/false, /*reuse_b=*/false);
}

// A fragment (16x32, bf16) from row-major bf16 matrix, leading dim ld.
// Lane layout: lanes 0-15 row M=lane, K = 0..15;
// lanes 16-31 row M=lane-16, K = 16..31 (matches B fragment).
__device__ __forceinline__ v16bf load_a_bf16(const bf16* base, int ld, int row0, int k0) {
    int lane = threadIdx.x & 31;
    int r  = row0 + (lane & 15);
    int kb = k0 + ((lane >> 4) << 4);
    v8bf lo = *(const v8bf*)(base + (size_t)r * ld + kb);
    v8bf hi = *(const v8bf*)(base + (size_t)r * ld + kb + 8);
    v16bf a;
#pragma unroll
    for (int i = 0; i < 8; ++i) { a[i] = lo[i]; a[8 + i] = hi[i]; }
    return a;
}

// Same A fragment but sourced from fp32 row-major data, converted to bf16.
__device__ __forceinline__ v16bf load_a_f32(const float* base, int ld, int row0, int k0) {
    int lane = threadIdx.x & 31;
    int r  = row0 + (lane & 15);
    int kb = k0 + ((lane >> 4) << 4);
    const float* p0 = base + (size_t)r * ld + kb;
    const float* p1 = p0 + 8;
    v16bf a;
#pragma unroll
    for (int i = 0; i < 8; ++i) { a[i] = (bf16)p0[i]; a[8 + i] = (bf16)p1[i]; }
    return a;
}

// B fragment (32x16, bf16) from K-major storage: BT[n][k], leading dim ld (K stride).
// Lanes 0-15: column N=lane, K=k0..k0+15 contiguous; lanes 16-31: K=k0+16..k0+31.
__device__ __forceinline__ v16bf load_b_bf16(const bf16* baseT, int ld, int n0, int k0) {
    int lane = threadIdx.x & 31;
    int n  = n0 + (lane & 15);
    int kb = k0 + ((lane >> 4) << 4);
    return *(const v16bf*)(baseT + (size_t)n * ld + kb);
}

// ---------------------------------------------------------------------------
// 1) Weight transpose + fp32 -> bf16 convert.  W:[rows,cols] -> WT:[cols,rows]
// ---------------------------------------------------------------------------
__global__ void wconv_kernel(const float* __restrict__ W, bf16* __restrict__ WT,
                             int rows, int cols) {
    int i = blockIdx.x * blockDim.x + threadIdx.x;
    if (i >= rows * cols) return;
    int c = i / rows, r = i % rows;
    WT[i] = (bf16)W[(size_t)r * cols + c];
}

// ---------------------------------------------------------------------------
// 2) Projection GEMM:  Y = (X @ W + b) * scale, reshaped per-head.
//    One wave = one 16x64 output tile (exactly one head wide):
//    A fragment loaded once per k-step, reused across 4 B tiles -> 4 WMMAs.
//      transpose_v==0 : Y[b][h][t][d]   (Q, K layout)
//      transpose_v==1 : Y[b][h][d][t]   (V^T layout for K-major B fragments)
// ---------------------------------------------------------------------------
__global__ __launch_bounds__(128)
void proj_kernel(const float* __restrict__ X, const bf16* __restrict__ WT,
                 const float* __restrict__ bias, bf16* __restrict__ Y,
                 float scale, int transpose_v) {
    int wave = (blockIdx.x * blockDim.x + threadIdx.x) >> 5;   // 4096 waves
    int rt = wave >> 3;                                        // 512 row tiles
    int ct = wave & 7;                                         // 8 head tiles
    int row0 = rt * 16, col0 = ct * 64;

    v8f acc0 = {}, acc1 = {}, acc2 = {}, acc3 = {};
#pragma unroll 4
    for (int k0 = 0; k0 < 512; k0 += 32) {
        v16bf a = load_a_f32(X, 512, row0, k0);           // shared across 4 tiles
        acc0 = wmma_bf16(a, load_b_bf16(WT, 512, col0 +  0, k0), acc0);
        acc1 = wmma_bf16(a, load_b_bf16(WT, 512, col0 + 16, k0), acc1);
        acc2 = wmma_bf16(a, load_b_bf16(WT, 512, col0 + 32, k0), acc2);
        acc3 = wmma_bf16(a, load_b_bf16(WT, 512, col0 + 48, k0), acc3);
    }

    int lane = threadIdx.x & 31;
    int n = lane & 15;
    int h = col0 >> 6;                                    // whole tile is one head
    v8f accs[4] = {acc0, acc1, acc2, acc3};
#pragma unroll
    for (int j = 0; j < 4; ++j) {
        int d  = j * 16 + n;
        float bv = bias[col0 + d];
#pragma unroll
        for (int r = 0; r < 8; ++r) {
            int m   = r + ((lane >> 4) << 3);
            int row = row0 + m;
            int bb = row >> 10, t = row & 1023;
            float y = (accs[j][r] + bv) * scale;
            if (transpose_v)
                Y[(((size_t)(bb * 8 + h) * 64) + d) * 1024 + t] = (bf16)y;
            else
                Y[(((size_t)(bb * 8 + h) * 1024) + t) * 64 + d] = (bf16)y;
        }
    }
}

// ---------------------------------------------------------------------------
// 3) Attention: one wave owns (b, h, 16-query stripe).
//    Pass1: S = Q K^T (pre-scaled Q), masked, into LDS (16x1024 f32).
//    Pass2: exact softmax stats via wave32 shuffles; exp() stored in place.
//    Pass3: per 32-key block -> normalized weights to HBM + bf16 stage tile
//           -> 4x WMMA against V^T, f32 accumulate; cv stored bf16.
//    Masks computed analytically (tril / 1-eye), saving 64MB of mask reads.
// ---------------------------------------------------------------------------
__global__ __launch_bounds__(128)
void attn_kernel(const bf16* __restrict__ Q,   // [B,H,T,64]
                 const bf16* __restrict__ K,   // [B,H,T,64]
                 const bf16* __restrict__ VT,  // [B,H,64,T]
                 float* __restrict__ AW,       // [B,H,T,T]
                 bf16* __restrict__ CV,        // [B,T,512]
                 int causal) {
    extern __shared__ char smem[];
    const int widx = threadIdx.x >> 5;
    const int lane = threadIdx.x & 31;
    float* S    = (float*)(smem + (size_t)widx * 66560);           // 16x1024 f32
    bf16* stage = (bf16*)(smem + (size_t)widx * 66560 + 65536);    // 16x32 bf16

    int wave = blockIdx.x * 4 + widx;    // 4096 waves total
    int b   = wave >> 9;
    int rem = wave & 511;
    int h   = rem >> 6;
    int qt  = rem & 63;

    const bf16* Qb = Q  + (((size_t)(b * 8 + h) * 1024) + qt * 16) * 64;
    const bf16* Kb = K  + ((size_t)(b * 8 + h) * 1024) * 64;
    const bf16* Vb = VT + ((size_t)(b * 8 + h) * 64) * 1024;
    float*     AWb = AW + (((size_t)(b * 8 + h) * 1024) + qt * 16) * 1024;

    // Q fragments reused across all 64 key tiles; warm V for pass 3.
    v16bf qa0 = load_a_bf16(Qb, 64, 0, 0);
    v16bf qa1 = load_a_bf16(Qb, 64, 0, 32);
    __builtin_prefetch(Vb + (size_t)lane * 2048, 0, 3);

    // ---- Pass 1: masked scores into LDS ----
    for (int kb = 0; kb < 64; ++kb) {
        if (kb < 63) __builtin_prefetch(Kb + (size_t)(kb + 1) * 16 * 64, 0, 3);
        v8f acc = {};
        acc = wmma_bf16(qa0, load_b_bf16(Kb, 64, kb * 16, 0),  acc);
        acc = wmma_bf16(qa1, load_b_bf16(Kb, 64, kb * 16, 32), acc);
        int n   = lane & 15;
        int key = kb * 16 + n;
#pragma unroll
        for (int r = 0; r < 8; ++r) {
            int m = r + ((lane >> 4) << 3);
            int q = qt * 16 + m;
            bool keep = causal ? (key <= q) : (key != q);
            S[m * 1024 + key] = keep ? acc[r] : NEG_BIG;
        }
    }
    asm volatile("s_wait_dscnt 0" ::: "memory");

    // ---- Pass 2: row max, exp in place, row sum (lane r keeps 1/sum of row r) ----
    float myinv = 1.0f;
    for (int r = 0; r < 16; ++r) {
        float mx = NEG_BIG;
        for (int i = lane; i < 1024; i += 32) mx = fmaxf(mx, S[r * 1024 + i]);
#pragma unroll
        for (int off = 16; off > 0; off >>= 1) mx = fmaxf(mx, __shfl_xor(mx, off, 32));
        float sum = 0.0f;
        for (int i = lane; i < 1024; i += 32) {
            float e = __expf(S[r * 1024 + i] - mx);
            S[r * 1024 + i] = e;
            sum += e;
        }
#pragma unroll
        for (int off = 16; off > 0; off >>= 1) sum += __shfl_xor(sum, off, 32);
        if (lane == r) myinv = 1.0f / sum;
    }
    asm volatile("s_wait_dscnt 0" ::: "memory");

    // ---- Pass 3: normalized weights out + P @ V (cv is 16x64) ----
    v8f cv0 = {}, cv1 = {}, cv2 = {}, cv3 = {};
    for (int kb = 0; kb < 32; ++kb) {
        int kc = kb * 32 + lane;
#pragma unroll
        for (int m = 0; m < 16; ++m) {
            float p = S[m * 1024 + kc] * __shfl(myinv, m, 32);
            AWb[(size_t)m * 1024 + kc] = p;         // coalesced fp32 weight store
            stage[m * 32 + lane] = (bf16)p;         // bf16 tile for WMMA A
        }
        asm volatile("s_wait_dscnt 0" ::: "memory"); // cross-lane LDS RAW
        v16bf pa = load_a_bf16(stage, 32, 0, 0);
        cv0 = wmma_bf16(pa, load_b_bf16(Vb, 1024,  0, kb * 32), cv0);
        cv1 = wmma_bf16(pa, load_b_bf16(Vb, 1024, 16, kb * 32), cv1);
        cv2 = wmma_bf16(pa, load_b_bf16(Vb, 1024, 32, kb * 32), cv2);
        cv3 = wmma_bf16(pa, load_b_bf16(Vb, 1024, 48, kb * 32), cv3);
    }

    bf16* CVb = CV + ((size_t)(b * 1024 + qt * 16)) * 512 + h * 64;
    int n = lane & 15;
#pragma unroll
    for (int r = 0; r < 8; ++r) {
        int m = r + ((lane >> 4) << 3);
        bf16* row = CVb + (size_t)m * 512;
        row[ 0 + n] = (bf16)cv0[r];
        row[16 + n] = (bf16)cv1[r];
        row[32 + n] = (bf16)cv2[r];
        row[48 + n] = (bf16)cv3[r];
    }
}

// ---------------------------------------------------------------------------
// 4) Output projection with fusion, 16x64 tile per wave:
//    OUT = (CVH @ Wo + bo) + 0.1 * tanh(CVF @ Wo + bo)
//    Each Wo B-fragment feeds both streams (8 WMMAs per 4 B + 2 A loads).
// ---------------------------------------------------------------------------
__global__ __launch_bounds__(128)
void oproj_kernel(const bf16* __restrict__ CVH, const bf16* __restrict__ CVF,
                  const bf16* __restrict__ WoT, const float* __restrict__ bo,
                  float* __restrict__ OUT) {
    int wave = (blockIdx.x * blockDim.x + threadIdx.x) >> 5;   // 4096 waves
    int rt = wave >> 3, ct = wave & 7;
    int row0 = rt * 16, col0 = ct * 64;

    v8f ah0 = {}, ah1 = {}, ah2 = {}, ah3 = {};
    v8f af0 = {}, af1 = {}, af2 = {}, af3 = {};
#pragma unroll 2
    for (int k0 = 0; k0 < 512; k0 += 32) {
        v16bf ha = load_a_bf16(CVH, 512, row0, k0);
        v16bf fa = load_a_bf16(CVF, 512, row0, k0);
        v16bf b0 = load_b_bf16(WoT, 512, col0 +  0, k0);
        ah0 = wmma_bf16(ha, b0, ah0);  af0 = wmma_bf16(fa, b0, af0);
        v16bf b1 = load_b_bf16(WoT, 512, col0 + 16, k0);
        ah1 = wmma_bf16(ha, b1, ah1);  af1 = wmma_bf16(fa, b1, af1);
        v16bf b2 = load_b_bf16(WoT, 512, col0 + 32, k0);
        ah2 = wmma_bf16(ha, b2, ah2);  af2 = wmma_bf16(fa, b2, af2);
        v16bf b3 = load_b_bf16(WoT, 512, col0 + 48, k0);
        ah3 = wmma_bf16(ha, b3, ah3);  af3 = wmma_bf16(fa, b3, af3);
    }

    int lane = threadIdx.x & 31;
    int n = lane & 15;
    v8f ahs[4] = {ah0, ah1, ah2, ah3};
    v8f afs[4] = {af0, af1, af2, af3};
#pragma unroll
    for (int j = 0; j < 4; ++j) {
        float bias = bo[col0 + j * 16 + n];
#pragma unroll
        for (int r = 0; r < 8; ++r) {
            int m = r + ((lane >> 4) << 3);
            float o = (ahs[j][r] + bias) + 0.1f * tanhf(afs[j][r] + bias);
            OUT[(size_t)(row0 + m) * 512 + col0 + j * 16 + n] = o;
        }
    }
}

// ---------------------------------------------------------------------------
// Launch
// ---------------------------------------------------------------------------
extern "C" void kernel_launch(void* const* d_in, const int* in_sizes, int n_in,
                              void* d_out, int out_size, void* d_ws, size_t ws_size,
                              hipStream_t stream) {
    const float* key_fwd   = (const float*)d_in[0];
    const float* value_fwd = (const float*)d_in[1];
    const float* query_fwd = (const float*)d_in[2];
    const float* key_bwd   = (const float*)d_in[3];
    const float* value_bwd = (const float*)d_in[4];
    const float* query_bwd = (const float*)d_in[5];
    // d_in[6]/d_in[7]: tgt_mask / identity_mask -- computed analytically on-chip.
    const float* Wk = (const float*)d_in[8];
    const float* bk = (const float*)d_in[9];
    const float* Wv = (const float*)d_in[10];
    const float* bv = (const float*)d_in[11];
    const float* Wq = (const float*)d_in[12];
    const float* bq = (const float*)d_in[13];
    const float* Wo = (const float*)d_in[14];
    const float* bo = (const float*)d_in[15];

    // Output layout (return order, fp32, flat):
    float* out      = (float*)d_out;
    float* out_cvf  = out;                       // [8,1024,512]
    float* out_cvb  = out + 4194304;             // [8,1024,512]
    float* aw_fh    = out + 8388608;             // [8,8,1024,1024]
    float* aw_ff    = aw_fh + 67108864;
    float* aw_bh    = aw_ff + 67108864;
    float* aw_bf    = aw_bh + 67108864;

    // Workspace carve-up (bf16 intermediates, 256B aligned blocks)
    char* ws = (char*)d_ws;
    size_t off = 0;
    bf16* WTk = (bf16*)(ws + off); off += 512 * 512 * sizeof(bf16);
    bf16* WTv = (bf16*)(ws + off); off += 512 * 512 * sizeof(bf16);
    bf16* WTq = (bf16*)(ws + off); off += 512 * 512 * sizeof(bf16);
    bf16* WoT = (bf16*)(ws + off); off += 512 * 512 * sizeof(bf16);
    const size_t headsz = (size_t)8 * 8 * 1024 * 64;          // elements
    bf16* Qf  = (bf16*)(ws + off); off += headsz * sizeof(bf16);
    bf16* Qb  = (bf16*)(ws + off); off += headsz * sizeof(bf16);
    bf16* Kf  = (bf16*)(ws + off); off += headsz * sizeof(bf16);
    bf16* Kb  = (bf16*)(ws + off); off += headsz * sizeof(bf16);
    bf16* VTf = (bf16*)(ws + off); off += headsz * sizeof(bf16);
    bf16* VTb = (bf16*)(ws + off); off += headsz * sizeof(bf16);
    const size_t cvsz = (size_t)8192 * 512;                   // elements
    bf16* cvfh = (bf16*)(ws + off); off += cvsz * sizeof(bf16);
    bf16* cvff = (bf16*)(ws + off); off += cvsz * sizeof(bf16);
    bf16* cvbh = (bf16*)(ws + off); off += cvsz * sizeof(bf16);
    bf16* cvbf = (bf16*)(ws + off); off += cvsz * sizeof(bf16);

    // 1) Weight transpose + bf16 convert (tiny)
    {
        dim3 g(1024), t(256);
        wconv_kernel<<<g, t, 0, stream>>>(Wk, WTk, 512, 512);
        wconv_kernel<<<g, t, 0, stream>>>(Wv, WTv, 512, 512);
        wconv_kernel<<<g, t, 0, stream>>>(Wq, WTq, 512, 512);
        wconv_kernel<<<g, t, 0, stream>>>(Wo, WoT, 512, 512);
    }

    // 2) Six projection GEMMs (1/sqrt(d_k)=0.125 folded exactly into Q)
    {
        dim3 g(1024), t(128);
        proj_kernel<<<g, t, 0, stream>>>(key_fwd,   WTk, bk, Kf,  1.0f,   0);
        proj_kernel<<<g, t, 0, stream>>>(value_fwd, WTv, bv, VTf, 1.0f,   1);
        proj_kernel<<<g, t, 0, stream>>>(query_fwd, WTq, bq, Qf,  0.125f, 0);
        proj_kernel<<<g, t, 0, stream>>>(key_bwd,   WTk, bk, Kb,  1.0f,   0);
        proj_kernel<<<g, t, 0, stream>>>(value_bwd, WTv, bv, VTb, 1.0f,   1);
        proj_kernel<<<g, t, 0, stream>>>(query_bwd, WTq, bq, Qb,  0.125f, 0);
    }

    // 3) Four attention passes (4 waves/block, 66560 B LDS per wave = 266240 B)
    {
        dim3 g(1024), t(128);
        size_t lds = 4 * 66560;
        attn_kernel<<<g, t, lds, stream>>>(Qf, Kf, VTf, aw_fh, cvfh, 1);
        attn_kernel<<<g, t, lds, stream>>>(Qf, Kb, VTb, aw_ff, cvff, 0);
        attn_kernel<<<g, t, lds, stream>>>(Qb, Kb, VTb, aw_bh, cvbh, 1);
        attn_kernel<<<g, t, lds, stream>>>(Qb, Kf, VTf, aw_bf, cvbf, 0);
    }

    // 4) Output projections with 0.1*tanh fusion
    {
        dim3 g(1024), t(128);
        oproj_kernel<<<g, t, 0, stream>>>(cvfh, cvff, WoT, bo, out_cvf);
        oproj_kernel<<<g, t, 0, stream>>>(cvbh, cvbf, WoT, bo, out_cvb);
    }
}
